// Head_20675972563638
// MI455X (gfx1250) — matrix operand order repaired and checked by reading, not compiled, not run on hardware
//
#include <hip/hip_runtime.h>
#include <hip/hip_bf16.h>

// Shapes (fixed by the reference): B=8, T=2048, C=1024, HS=128
#define BB  8
#define TT  2048
#define CC  1024
#define HSZ 128

typedef _Float16 f16;
typedef __attribute__((ext_vector_type(16))) _Float16 v16h;
typedef __attribute__((ext_vector_type(8)))  _Float16 v8h;
typedef __attribute__((ext_vector_type(8)))  float    v8f;
typedef __attribute__((ext_vector_type(4)))  float    v4f;
typedef __attribute__((ext_vector_type(4)))  unsigned int v4u;
typedef __attribute__((ext_vector_type(8)))  int      v8i;
typedef __attribute__((ext_vector_type(4)))  int      v4i;

#if __has_builtin(__builtin_amdgcn_tensor_load_to_lds) && __has_builtin(__builtin_amdgcn_s_wait_tensorcnt)
#define USE_TDM 1
#else
#define USE_TDM 0
#endif

union V16U { v16h v; v8h h[2]; };

// ---- WMMA wrapper: D = A(16x32 f16) * B(32x16 f16) + C(16x16 f32) ----
__device__ __forceinline__ v8f wmma_f16(v16h a, v16h b, v8f c) {
  return __builtin_amdgcn_wmma_f32_16x16x32_f16(
      /*neg_a=*/false, a, /*neg_b=*/false, b,
      /*c_mod=*/(short)0, c, /*reuse_a=*/false, /*reuse_b=*/false);
}

// A-fragment (16x32, MxK): lane L holds row L%16, K = khalf*16..+15
// contiguous (two 16-byte loads), matching the B-fragment K order.
__device__ __forceinline__ v16h load_a_frag(const f16* rowbase, int khalf) {
  V16U u;
  u.h[0] = *(const v8h*)(rowbase + khalf * 16);
  u.h[1] = *(const v8h*)(rowbase + khalf * 16 + 8);
  return u.v;
}

// B-fragment (32x16, KxN): lane L holds column L%16; lanes 0-15 K=0..15,
// lanes 16-31 K=16..31 (one contiguous 32-byte chunk along K).
__device__ __forceinline__ v16h load_b_frag(const f16* colbase, int khalf) {
  return *(const v16h*)(colbase + khalf * 16);
}

__device__ __forceinline__ v8f vzero8() {
  return (v8f){0.f,0.f,0.f,0.f,0.f,0.f,0.f,0.f};
}

#if USE_TDM
// LDS byte offset = low 32 bits of the generic (flat) shared-memory address.
__device__ __forceinline__ unsigned int lds_off_of(const void* p) {
  return (unsigned int)(uintptr_t)p;
}

// TDM: contiguous 1D copy of 8192 bytes (K-tile), data_size=8B, tile_dim0=1024.
__device__ __forceinline__ void tdm_load_1d_8k(unsigned int lds, const f16* g) {
  unsigned long long ga = (unsigned long long)g;
  v4u g0 = { 1u,                                   // count=1, user mode
             lds,                                  // lds_addr
             (unsigned int)ga,                     // global_addr[31:0]
             (unsigned int)((ga >> 32) & 0x01FFFFFFu) | (2u << 30) }; // addr[56:32] | type=2
  // group1: data_size=3(8B); tensor_dim0=1<<20; tensor_dim1=1; tile_dim0=1024;
  //         tile_dim1=0(unused); tensor_dim0_stride=1024
  v8i g1 = { (int)(3u << 16),
             0,
             (int)(16u | (1u << 16)),              // dim0>>16=16 | dim1.lo<<16
             (int)(1024u << 16),                   // dim1>>16=0 | tile_dim0<<16
             0,                                    // tile_dim1=0, tile_dim2=0
             1024,                                 // dim0_stride.lo
             0, 0 };
  v4i z4 = {0, 0, 0, 0};
  v8i z8 = {0, 0, 0, 0, 0, 0, 0, 0};
  __builtin_amdgcn_tensor_load_to_lds(g0, g1, z4, z4, z8, 0);
}

// TDM: 2D tile, 128 rows x 32 f16 elements, row stride `stride` elements.
// Used for the V-tile (stride=TT) and the W-chunk (stride=CC).
__device__ __forceinline__ void tdm_load_2d_128x32(unsigned int lds, const f16* g,
                                                   unsigned int stride) {
  unsigned long long ga = (unsigned long long)g;
  v4u g0 = { 1u, lds, (unsigned int)ga,
             (unsigned int)((ga >> 32) & 0x01FFFFFFu) | (2u << 30) };
  // data_size=1(2B); tensor_dim0=stride; tensor_dim1=65536; tile_dim0=32;
  // tile_dim1=128; tensor_dim0_stride=stride
  v8i g1 = { (int)(1u << 16),
             (int)(stride << 16),                  // dim0.lo<<16
             (int)(stride >> 16),                  // dim0.hi | dim1.lo(0)<<16
             (int)(1u | (32u << 16)),              // dim1>>16=1 | tile_dim0<<16
             128,                                  // tile_dim1=128, tile_dim2=0
             (int)stride,                          // dim0_stride.lo
             0, 0 };
  v4i z4 = {0, 0, 0, 0};
  v8i z8 = {0, 0, 0, 0, 0, 0, 0, 0};
  __builtin_amdgcn_tensor_load_to_lds(g0, g1, z4, z4, z8, 0);
}
#endif // USE_TDM

// ------------------- Kernel 1: x fp32 -> f16 (vectorized) -------------------
__global__ void cvt_x_kernel(const float* __restrict__ x, f16* __restrict__ xh, int n8) {
  int i = blockIdx.x * blockDim.x + threadIdx.x;
  if (i >= n8) return;
  const v4f* xp = (const v4f*)x + (size_t)i * 2;
  v4f a = xp[0], b = xp[1];
  v8h o;
  o[0]=(f16)a[0]; o[1]=(f16)a[1]; o[2]=(f16)a[2]; o[3]=(f16)a[3];
  o[4]=(f16)b[0]; o[5]=(f16)b[1]; o[6]=(f16)b[2]; o[7]=(f16)b[3];
  *((v8h*)xh + i) = o;
}

// ---------- Kernel 2: W [C][HS] fp32 -> Wt [HS][C] f16 (x3) ----------
__global__ void cvt_w_kernel(const float* __restrict__ Wq, const float* __restrict__ Wk,
                             const float* __restrict__ Wv,
                             f16* __restrict__ WtQ, f16* __restrict__ WtK, f16* __restrict__ WtV) {
  int i = blockIdx.x * blockDim.x + threadIdx.x;
  if (i >= CC * HSZ) return;
  int c = i / HSZ, h = i % HSZ;
  WtQ[h * CC + c] = (f16)Wq[i];
  WtK[h * CC + c] = (f16)Wk[i];
  WtV[h * CC + c] = (f16)Wv[i];
}

// ---------- projection helper: acc[0..7] = x_tile(16xC) * W(Cx128) ----------
#if USE_TDM
// TDM variant: the 128x32 W-chunk for step kk is shared by all 4 waves, so
// wave 0 double-buffers it into LDS while everyone computes step kk-1.
__device__ __forceinline__ void proj_tile(const f16* __restrict__ xrow,
                                          const f16* __restrict__ Wt,
                                          f16 (*wbuf)[HSZ * 32], bool issuer,
                                          int row, int khalf, v8f acc[8]) {
#pragma unroll
  for (int nt = 0; nt < 8; ++nt) acc[nt] = vzero8();
  if (issuer) tdm_load_2d_128x32(lds_off_of(&wbuf[0][0]), Wt, CC);
  for (int kk = 0; kk < 32; ++kk) {              // C = 32 * 32
    const int cur = kk & 1;
    if (issuer) {
      if (kk + 1 < 32) {
        tdm_load_2d_128x32(lds_off_of(&wbuf[cur ^ 1][0]), Wt + (kk + 1) * 32, CC);
        __builtin_amdgcn_s_wait_tensorcnt(1);    // chunk kk complete (in-order)
      } else {
        __builtin_amdgcn_s_wait_tensorcnt(0);
      }
    }
    __syncthreads();                             // publish chunk kk
    v16h a = load_a_frag(xrow + kk * 32, khalf); // 2 global loads / 8 WMMAs
    const f16* wp = &wbuf[cur][0];               // [128 hs][32 k]
    v16h b[8];
#pragma unroll
    for (int nt = 0; nt < 8; ++nt)
      b[nt] = load_b_frag(wp + (size_t)(nt * 16 + row) * 32, khalf);
#pragma unroll
    for (int nt = 0; nt < 8; ++nt)
      acc[nt] = wmma_f16(a, b[nt], acc[nt]);
    __syncthreads();                             // done with chunk kk buffer
  }
}
#else
// Fallback: direct global loads, skewed two WMMAs ahead.
__device__ __forceinline__ void proj_tile(const f16* __restrict__ xrow,
                                          const f16* __restrict__ Wt,
                                          int row, int khalf, v8f acc[8]) {
#pragma unroll
  for (int nt = 0; nt < 8; ++nt) acc[nt] = vzero8();
  for (int kk = 0; kk < 32; ++kk) {
    v16h a = load_a_frag(xrow + kk * 32, khalf);
    v16h b[8];
    b[0] = load_b_frag(Wt + (size_t)(0 * 16 + row) * CC + kk * 32, khalf);
    b[1] = load_b_frag(Wt + (size_t)(1 * 16 + row) * CC + kk * 32, khalf);
#pragma unroll
    for (int nt = 0; nt < 8; ++nt) {
      if (nt + 2 < 8)
        b[nt + 2] = load_b_frag(Wt + (size_t)((nt + 2) * 16 + row) * CC + kk * 32, khalf);
      acc[nt] = wmma_f16(a, b[nt], acc[nt]);
    }
  }
}
#endif

// ---------- Kernel 3: QKV projection (WMMA) ----------
__global__ __launch_bounds__(128) void qkv_kernel(
    const f16* __restrict__ xh,
    const f16* __restrict__ WtQ, const f16* __restrict__ WtK, const f16* __restrict__ WtV,
    f16* __restrict__ qh, f16* __restrict__ kh, f16* __restrict__ vT) {
#if USE_TDM
  __shared__ f16 wbuf[2][HSZ * 32];              // 2 x 8 KB shared W chunk
#endif
  const int wid  = threadIdx.x >> 5;
  const int lane = threadIdx.x & 31;
  const int row  = lane & 15;
  const int khalf = lane >> 4;
  const int rt = blockIdx.x * 4 + wid;           // 16-row tile index over B*T

  const f16* xrow = xh + (size_t)(rt * 16 + row) * CC;
  v8f acc[8];
#if USE_TDM
  const bool issuer = (wid == 0);
#define PROJ(W) proj_tile(xrow, (W), wbuf, issuer, row, khalf, acc)
#else
#define PROJ(W) proj_tile(xrow, (W), row, khalf, acc)
#endif

  // ---- Q ----
  PROJ(WtQ);
#pragma unroll
  for (int nt = 0; nt < 8; ++nt)
#pragma unroll
    for (int r = 0; r < 8; ++r) {
      int m = (lane < 16) ? r : (r + 8);
      qh[(size_t)(rt * 16 + m) * HSZ + nt * 16 + row] = (f16)acc[nt][r];
    }

  // ---- K ----
  PROJ(WtK);
#pragma unroll
  for (int nt = 0; nt < 8; ++nt)
#pragma unroll
    for (int r = 0; r < 8; ++r) {
      int m = (lane < 16) ? r : (r + 8);
      kh[(size_t)(rt * 16 + m) * HSZ + nt * 16 + row] = (f16)acc[nt][r];
    }

  // ---- V (stored transposed: vT[B][HS][T]) ----
  PROJ(WtV);
#pragma unroll
  for (int nt = 0; nt < 8; ++nt)
#pragma unroll
    for (int r = 0; r < 8; ++r) {
      int m = (lane < 16) ? r : (r + 8);
      int g = rt * 16 + m;
      int b_ = g / TT, t = g % TT;
      vT[((size_t)b_ * HSZ + nt * 16 + row) * TT + t] = (f16)acc[nt][r];
    }
#undef PROJ
}

// ---------- Kernel 4: flash attention ----------
// 4 waves/block, all on the same batch. With TDM: wave 0 double-buffers the
// K-tile (32x128 f16, contiguous) and V-tile (128x32 f16, strided) into LDS
// via tensor_load_to_lds; TENSORcnt in-order completion lets it wait for the
// current tile while the next pair is in flight. All waves share the tiles.
__global__ __launch_bounds__(128) void attn_kernel(
    const f16* __restrict__ qh, const f16* __restrict__ kh, const f16* __restrict__ vT,
    const int* __restrict__ mask, float* __restrict__ out) {
#if USE_TDM
  __shared__ f16 kbuf[2][32 * HSZ];              // 2 x 8 KB
  __shared__ f16 vbuf[2][HSZ * 32];              // 2 x 8 KB
#endif
  __shared__ f16 pbuf[4][16 * 32];               // per-wave P staging tile

  const int wid  = threadIdx.x >> 5;
  const int lane = threadIdx.x & 31;
  const int row  = lane & 15;
  const int khalf = lane >> 4;
  const int qt = blockIdx.x * 4 + wid;           // global 16-query tile over B*T
  const int b = qt / (TT / 16);                  // same for all 4 waves in block

  // Q: 4 A-fragments kept in registers
  const f16* qrow = qh + (size_t)(qt * 16 + row) * HSZ;
  v16h qa[4];
#pragma unroll
  for (int kk = 0; kk < 4; ++kk) qa[kk] = load_a_frag(qrow + kk * 32, khalf);

  v8f o[8];
#pragma unroll
  for (int nt = 0; nt < 8; ++nt) o[nt] = vzero8();
  float mrow[8], lrow[8];
#pragma unroll
  for (int r = 0; r < 8; ++r) { mrow[r] = -1e30f; lrow[r] = 0.f; }

  const float scale = 0.08838834764831845f;      // 1/sqrt(128)
  f16* myp = &pbuf[wid][0];
  const f16* kmat  = kh + (size_t)b * TT * HSZ;  // [T][HS]
  const f16* vbase = vT + (size_t)b * HSZ * TT;  // [HS][T]
  const int* mb    = mask + b * TT;

#if USE_TDM
  const bool issuer = (wid == 0);
  if (issuer) {
    tdm_load_1d_8k(lds_off_of(&kbuf[0][0]), kmat);
    tdm_load_2d_128x32(lds_off_of(&vbuf[0][0]), vbase, TT);
  }
#endif

  for (int kt = 0; kt < TT / 32; ++kt) {
    const int key0 = kt * 32;

#if USE_TDM
    const int cur = kt & 1;
    if (issuer) {
      if (kt + 1 < TT / 32) {
        tdm_load_1d_8k(lds_off_of(&kbuf[cur ^ 1][0]), kmat + (size_t)(key0 + 32) * HSZ);
        tdm_load_2d_128x32(lds_off_of(&vbuf[cur ^ 1][0]), vbase + key0 + 32, TT);
        __builtin_amdgcn_s_wait_tensorcnt(2);    // pair kt complete (in-order)
      } else {
        __builtin_amdgcn_s_wait_tensorcnt(0);
      }
    }
    __syncthreads();                             // publish tile kt to all waves
    const f16* kp = &kbuf[cur][0];               // [32 keys][128 hs]
    const f16* vp = &vbuf[cur][0];               // [128 hs][32 keys]
    const int VS = 32;
#else
    if (kt + 1 < TT / 32) {
      __builtin_prefetch(kmat + (size_t)(key0 + 32 + row) * HSZ, 0, 1);
      __builtin_prefetch(kmat + (size_t)(key0 + 48 + row) * HSZ, 0, 1);
      __builtin_prefetch(vbase + (size_t)(row * 8) * TT + key0 + 32, 0, 1);
    }
    const f16* kp = kmat + (size_t)key0 * HSZ;
    const f16* vp = vbase + key0;
    const int VS = TT;
#endif

    // S = Q * K^T for 16 queries x 32 keys (two 16x16 tiles)
    v8f s0 = vzero8();
    v8f s1 = vzero8();
    {
      v16h bk[4];
#pragma unroll
      for (int kk = 0; kk < 4; ++kk)
        bk[kk] = load_b_frag(kp + (size_t)(row) * HSZ + kk * 32, khalf);
#pragma unroll
      for (int kk = 0; kk < 4; ++kk) s0 = wmma_f16(qa[kk], bk[kk], s0);
#pragma unroll
      for (int kk = 0; kk < 4; ++kk)
        bk[kk] = load_b_frag(kp + (size_t)(16 + row) * HSZ + kk * 32, khalf);
#pragma unroll
      for (int kk = 0; kk < 4; ++kk) s1 = wmma_f16(qa[kk], bk[kk], s1);
    }

    // scale + key mask (column = lane%16 within each tile)
    const float madd0 = mb[key0 + row]      ? 0.f : -1e30f;
    const float madd1 = mb[key0 + 16 + row] ? 0.f : -1e30f;
    float p0[8], p1[8];
#pragma unroll
    for (int r = 0; r < 8; ++r) {
      p0[r] = s0[r] * scale + madd0;
      p1[r] = s1[r] * scale + madd1;
    }

    // online softmax; row stats and O accumulators share the same
    // (VGPR slot, lane-half) -> row mapping, so rescale is local.
#pragma unroll
    for (int r = 0; r < 8; ++r) {
      float mx = fmaxf(p0[r], p1[r]);
      mx = fmaxf(mx, __shfl_xor(mx, 8, 32));
      mx = fmaxf(mx, __shfl_xor(mx, 4, 32));
      mx = fmaxf(mx, __shfl_xor(mx, 2, 32));
      mx = fmaxf(mx, __shfl_xor(mx, 1, 32));
      float mnew = fmaxf(mrow[r], mx);
      float fac  = __expf(mrow[r] - mnew);
      mrow[r] = mnew;
      float e0 = __expf(p0[r] - mnew);
      float e1 = __expf(p1[r] - mnew);
      float rs = e0 + e1;
      rs += __shfl_xor(rs, 8, 32);
      rs += __shfl_xor(rs, 4, 32);
      rs += __shfl_xor(rs, 2, 32);
      rs += __shfl_xor(rs, 1, 32);
      lrow[r] = lrow[r] * fac + rs;
#pragma unroll
      for (int nt = 0; nt < 8; ++nt) o[nt][r] *= fac;
      int m = (lane < 16) ? r : (r + 8);
      myp[m * 32 + row]      = (f16)e0;
      myp[m * 32 + 16 + row] = (f16)e1;
    }

    // reload P as an A fragment (16x32); DS ops are in-order per wave
    v16h pa = load_a_frag(myp + row * 32, khalf);

    // O += P * V
    {
      v16h bv[4];
#pragma unroll
      for (int nt = 0; nt < 4; ++nt)
        bv[nt] = load_b_frag(vp + (size_t)(nt * 16 + row) * VS, khalf);
#pragma unroll
      for (int nt = 0; nt < 4; ++nt) o[nt] = wmma_f16(pa, bv[nt], o[nt]);
#pragma unroll
      for (int nt = 0; nt < 4; ++nt)
        bv[nt] = load_b_frag(vp + (size_t)((nt + 4) * 16 + row) * VS, khalf);
#pragma unroll
      for (int nt = 0; nt < 4; ++nt) o[nt + 4] = wmma_f16(pa, bv[nt], o[nt + 4]);
    }

#if USE_TDM
    __syncthreads();   // all waves done with buffer kt before it is re-filled
#endif
  }

  // epilogue: out[B*T][HS] fp32
#pragma unroll
  for (int nt = 0; nt < 8; ++nt) {
#pragma unroll
    for (int r = 0; r < 8; ++r) {
      int m = (lane < 16) ? r : (r + 8);
      out[(size_t)(qt * 16 + m) * HSZ + nt * 16 + row] = o[nt][r] / lrow[r];
    }
  }
}

extern "C" void kernel_launch(void* const* d_in, const int* in_sizes, int n_in,
                              void* d_out, int out_size, void* d_ws, size_t ws_size,
                              hipStream_t stream) {
  const float* x    = (const float*)d_in[0];
  const int*   msk  = (const int*)d_in[1];
  const float* Wk   = (const float*)d_in[2];
  const float* Wq   = (const float*)d_in[3];
  const float* Wv   = (const float*)d_in[4];
  float* out = (float*)d_out;

  // workspace carving
  char* ws = (char*)d_ws;
  size_t off = 0;
  f16* xh  = (f16*)(ws + off); off += (size_t)BB * TT * CC * sizeof(f16);   // 32 MB
  f16* qh  = (f16*)(ws + off); off += (size_t)BB * TT * HSZ * sizeof(f16);  // 4 MB
  f16* kh  = (f16*)(ws + off); off += (size_t)BB * TT * HSZ * sizeof(f16);  // 4 MB
  f16* vT  = (f16*)(ws + off); off += (size_t)BB * TT * HSZ * sizeof(f16);  // 4 MB
  f16* WtQ = (f16*)(ws + off); off += (size_t)CC * HSZ * sizeof(f16);
  f16* WtK = (f16*)(ws + off); off += (size_t)CC * HSZ * sizeof(f16);
  f16* WtV = (f16*)(ws + off); off += (size_t)CC * HSZ * sizeof(f16);
  (void)ws_size;

  {
    int n8 = (BB * TT * CC) / 8;
    cvt_x_kernel<<<(n8 + 255) / 256, 256, 0, stream>>>(x, xh, n8);
  }
  {
    int n = CC * HSZ;
    cvt_w_kernel<<<(n + 255) / 256, 256, 0, stream>>>(Wq, Wk, Wv, WtQ, WtK, WtV);
  }
  {
    int tiles = (BB * TT) / 16;                  // 1024
    qkv_kernel<<<tiles / 4, 128, 0, stream>>>(xh, WtQ, WtK, WtV, qh, kh, vT);
  }
  {
    int tiles = (BB * TT) / 16;                  // 1024
    attn_kernel<<<tiles / 4, 128, 0, stream>>>(qh, kh, vT, msk, out);
  }
}
